// ResIERes_44719199486515
// MI455X (gfx1250) — matrix operand order repaired and checked by reading, not compiled, not run on hardware
//
#include <hip/hip_runtime.h>
#include <hip/hip_bf16.h>
#include <math.h>

// ---------------------------------------------------------------------------
// MI455X (gfx1250) implementation.
// Memory-bound workload (~GB moved vs ~13 GFLOP of conv math): convs are
// implicit-GEMM WMMA (v_wmma_f32_16x16x32_f16, f32 accum) with branchless
// b128 LDS fragment loads and fused epilogues. The 8-channel intermediate
// activations are stored as f16 (they are consumed as f16 by WMMA anyway),
// halving the dominant HBM traffic at 23.3 TB/s. All global stats are
// deterministic LDS tree reductions in f32.
// ---------------------------------------------------------------------------

#define NB 8
#define HH 512
#define WW 512
#define HW (HH * WW)          // 262144 = 1 << 18

typedef __attribute__((ext_vector_type(16))) _Float16 v16h;
typedef __attribute__((ext_vector_type(8)))  _Float16 v8h;
typedef __attribute__((ext_vector_type(8)))  float    v8f;

__device__ __forceinline__ float sigmoidf_(float x) { return 1.f / (1.f + __expf(-x)); }
__device__ __forceinline__ float satf_(float x) { return fminf(fmaxf(x, 0.f), 1.f); }
__device__ __forceinline__ float log_approx_f(float x) {
  x = fmaxf(x, 1e-6f);
  float t = x - 1.f;
  return 0.69f * t - 0.33f * t * t;
}

// ---------------------------------------------------------------------------
// Implicit-GEMM 3x3 SAME conv via WMMA, K reordered as k = tap*8 + cin with
// channel padding to 8 and tap padding to 12 (K = 96, 3 k-steps), so every
// A/B fragment is assembled from aligned, guard-free ds_load_b128.
//
//   A (16 x 96) : weights wl[16][104] f16, zero padded; row stride 104 f16
//                 (52 dwords) -> 16 lanes' b128 reads tile all 64 LDS banks.
//   B (96 x 16) : input tile[4][132][8] f16, pixel-major/channel-minor; the
//                 dy=3 plane stays zero for the 3 padded taps.
//   C (16 x 16) : f32 accumulators.
//
// One wave = 16 consecutive pixels of one row x all out channels.
// One block (8 waves) = 128 pixels of one row.
// TIN/TOUT: float or _Float16 activation storage.
// ACT: 0 = none, 1 = relu, 2 = +residual(f32) then sigmoid
// ---------------------------------------------------------------------------
template <typename TIN, typename TOUT, int CIN, int COUT, int ACT>
__global__ __launch_bounds__(256) void conv3x3_wmma_k(
    const TIN* __restrict__ in, const float* __restrict__ wgt,
    const float* __restrict__ bias, const float* __restrict__ res,
    TOUT* __restrict__ out) {
  __shared__ __align__(16) _Float16 tile[4][132][8];   // 8448 B
  __shared__ __align__(16) _Float16 wl[16][104];       // 3328 B

  const int tid = threadIdx.x;
  const int bw  = blockIdx.x & 3;                 // W / 128 = 4 tiles per row
  const int row = (blockIdx.x >> 2) & (HH - 1);
  const int n   = blockIdx.x >> 11;               // / (4*512)
  const int x0  = bw * 128;

  // ---- zero-fill both LDS arrays (dword stores) ----
  {
    unsigned* t32 = (unsigned*)&tile[0][0][0];
    for (int i = tid; i < (4 * 132 * 8) / 2; i += 256) t32[i] = 0u;
    unsigned* w32 = (unsigned*)&wl[0][0];
    for (int i = tid; i < (16 * 104) / 2; i += 256) w32[i] = 0u;
  }
  __syncthreads();

  // ---- stage weights: wl[m][tap*8+cin] = wgt[m][cin][tap] ----
  for (int i = tid; i < COUT * CIN * 9; i += 256) {
    int m   = i / (CIN * 9);
    int r   = i - m * (CIN * 9);
    int cin = r / 9;
    int tap = r - cin * 9;
    wl[m][tap * 8 + cin] = (_Float16)wgt[i];
  }

  // ---- stage input halo tile: tile[dy][xx][cin], zero padded ----
  for (int i = tid; i < 3 * 130 * CIN; i += 256) {
    int xx = i % 130;
    int r2 = i / 130;
    int c  = r2 % CIN;
    int dy = r2 / CIN;
    int gx = x0 + xx - 1;
    int gy = row + dy - 1;
    _Float16 v = (_Float16)0.f;
    if ((unsigned)gx < (unsigned)WW && (unsigned)gy < (unsigned)HH)
      v = (_Float16)in[((size_t)(n * CIN + c) * HH + gy) * WW + gx];
    tile[dy][xx][c] = v;
  }
  // Speculative prefetch of an upcoming row's input (global_prefetch_b8).
  if (tid < CIN * 2) {
    int c  = tid >> 1;
    int hx = (tid & 1) * 64;
    __builtin_prefetch(&in[((size_t)(n * CIN + c) * HH + ((row + 2) & (HH - 1))) * WW + x0 + hx], 0, 1);
  }
  __syncthreads();

  const int wave = tid >> 5;
  const int lane = tid & 31;
  const int col  = lane & 15;         // pixel column N / A-matrix row M
  const int half = lane >> 4;
  const int px   = wave * 16 + col;   // tile-local output pixel

  v8f acc = {};
#pragma unroll
  for (int kt = 0; kt < 3; ++kt) {
    // A fragment: one 16-run of K (taps kt*4+2*half and kt*4+2*half+1).
    const v8h a0 = *(const v8h*)&wl[col][(kt * 4 + 2 * half) * 8];
    const v8h a1 = *(const v8h*)&wl[col][(kt * 4 + 2 * half + 1) * 8];
    const v16h a = __builtin_shufflevector(a0, a1, 0, 1, 2, 3, 4, 5, 6, 7,
                                           8, 9, 10, 11, 12, 13, 14, 15);
    // B fragment: one 16-run of K = taps t0, t0+1 for this lane's pixel.
    const int t0  = kt * 4 + half * 2;
    const int t1  = t0 + 1;
    const int dy0 = t0 / 3, dx0 = t0 - dy0 * 3;
    const int dy1 = t1 / 3, dx1 = t1 - dy1 * 3;
    const v8h b0 = *(const v8h*)&tile[dy0][px + dx0][0];
    const v8h b1 = *(const v8h*)&tile[dy1][px + dx1][0];
    const v16h b = __builtin_shufflevector(b0, b1, 0, 1, 2, 3, 4, 5, 6, 7,
                                           8, 9, 10, 11, 12, 13, 14, 15);
    acc = __builtin_amdgcn_wmma_f32_16x16x32_f16(false, a, false, b,
                                                 (short)0, acc, false, false);
  }

  // C/D layout: VGPR r, lanes 0-15 -> (M=r, N=lane); lanes 16-31 -> (M=8+r, N=lane-16)
  const int gx = x0 + px;
#pragma unroll
  for (int r = 0; r < 8; ++r) {
    int ch = half * 8 + r;
    if (ch < COUT) {
      float v = acc[r] + bias[ch];
      if (ACT == 1) v = fmaxf(v, 0.f);
      if (ACT == 2) {
        v += res[((size_t)(n * COUT + ch) * HH + row) * WW + gx];
        v = sigmoidf_(v);
      }
      out[((size_t)(n * COUT + ch) * HH + row) * WW + gx] = (TOUT)v;
    }
  }
}

// ---------------------------------------------------------------------------
// Deterministic per-(n,c) sum and sum-of-squares over H*W. One block per (n,c).
// ---------------------------------------------------------------------------
template <typename T>
__global__ __launch_bounds__(256) void reduce_stats_k(
    const T* __restrict__ in, float* __restrict__ sums, float* __restrict__ sqs) {
  __shared__ float ss[256], s2[256];
  const int b = blockIdx.x;
  const T* p = in + (size_t)b * HW;
  float s = 0.f, q = 0.f;
  for (int i = threadIdx.x; i < HW; i += 256) {
    float v = (float)p[i];
    s += v;
    q += v * v;
  }
  ss[threadIdx.x] = s;
  s2[threadIdx.x] = q;
  __syncthreads();
  for (int off = 128; off > 0; off >>= 1) {
    if (threadIdx.x < off) {
      ss[threadIdx.x] += ss[threadIdx.x + off];
      s2[threadIdx.x] += s2[threadIdx.x + off];
    }
    __syncthreads();
  }
  if (threadIdx.x == 0) { sums[b] = ss[0]; sqs[b] = s2[0]; }
}

// ---------------------------------------------------------------------------
// Channel-attention squeeze-excite: per image, 8 means -> relu(1x1 8->2)
// -> sigmoid(1x1 2->8) channel scales.
// ---------------------------------------------------------------------------
__global__ void ca_scale_k(const float* __restrict__ sums,
                           const float* __restrict__ w1, const float* __restrict__ b1,
                           const float* __restrict__ w2, const float* __restrict__ b2,
                           float* __restrict__ cs) {
  int n = blockIdx.x;
  int c = threadIdx.x;
  if (c >= 8) return;
  float m[8];
#pragma unroll
  for (int i = 0; i < 8; ++i) m[i] = sums[n * 8 + i] * (1.f / HW);
  float y0 = b1[0], y1 = b1[1];
#pragma unroll
  for (int i = 0; i < 8; ++i) { y0 += w1[i] * m[i]; y1 += w1[8 + i] * m[i]; }
  y0 = fmaxf(y0, 0.f);
  y1 = fmaxf(y1, 0.f);
  float z = w2[c * 2 + 0] * y0 + w2[c * 2 + 1] * y1 + b2[c];
  cs[n * 8 + c] = sigmoidf_(z);
}

// ---------------------------------------------------------------------------
// cp_layer apply + attention residual (f16 activations):
//   a_c = x_c^2 * cs_c  (CA);  p = sigmoid(pw2 . relu(pw1 . a + pb1) + pb2)
//   out_c = a_c^2 * p + h0_c
// ---------------------------------------------------------------------------
__global__ __launch_bounds__(256) void cp_apply_k(
    const _Float16* __restrict__ t2, const _Float16* __restrict__ h0,
    const float* __restrict__ cs,
    const float* __restrict__ pw1, const float* __restrict__ pb1,
    const float* __restrict__ pw2, const float* __restrict__ pb2,
    _Float16* __restrict__ outb) {
  int idx = blockIdx.x * 256 + threadIdx.x;   // n*HW + p
  int n = idx >> 18;
  int p = idx & (HW - 1);
  float a[8];
#pragma unroll
  for (int c = 0; c < 8; ++c) {
    float v = (float)t2[(size_t)(n * 8 + c) * HW + p];
    a[c] = v * v * cs[n * 8 + c];
  }
  float y0 = pb1[0], y1 = pb1[1];
#pragma unroll
  for (int c = 0; c < 8; ++c) { y0 += pw1[c] * a[c]; y1 += pw1[8 + c] * a[c]; }
  y0 = fmaxf(y0, 0.f);
  y1 = fmaxf(y1, 0.f);
  float ps = sigmoidf_(pw2[0] * y0 + pw2[1] * y1 + pb2[0]);
#pragma unroll
  for (int c = 0; c < 8; ++c) {
    size_t o = (size_t)(n * 8 + c) * HW + p;
    outb[o] = (_Float16)(a[c] * a[c] * ps + (float)h0[o]);
  }
}

// ---------------------------------------------------------------------------
// White balance coefficients (scalar work, 1 thread).
// wb[n*8+{0..5}] = {Gm-Rm, Gm-Bm, Bm-Rm, alpha, beta, gamma}; flags = {c1,c3,any}
// ---------------------------------------------------------------------------
__global__ void wb_coeff_k(const float* __restrict__ sums,
                           float* __restrict__ wb, float* __restrict__ flags) {
  if (threadIdx.x != 0 || blockIdx.x != 0) return;
  float Rm[NB], Gm[NB], Bm[NB];
  float gb = 0.f, rb = 0.f, bg = 0.f;
  int anyc = 0;
  for (int n = 0; n < NB; ++n) {
    Rm[n] = sums[n * 3 + 0] * (1.f / HW);
    Gm[n] = sums[n * 3 + 1] * (1.f / HW);
    Bm[n] = sums[n * 3 + 2] * (1.f / HW);
    float K = Rm[n] + Gm[n] + Bm[n];
    if (Gm[n] * 3.f > 2.f * K) anyc = 1;
    gb += Gm[n] - Bm[n];
    rb += Rm[n] - Bm[n];
    bg += Bm[n] - Gm[n];
  }
  gb *= (1.f / NB); rb *= (1.f / NB); bg *= (1.f / NB);
  int c1 = gb >= 0.1f, c2 = rb <= 0.1f, c3 = bg > 0.1f;
  for (int n = 0; n < NB; ++n) {
    float alpha = 1.f - log_approx_f(fmaxf(Gm[n] - Rm[n], 1e-6f));
    float beta0 = -log_approx_f(fmaxf(Gm[n] - Bm[n], 1e-6f));
    float beta  = c2 ? 1.f / (1.f + beta0) : 1.f + beta0;
    if (c1 && !c2) alpha = 0.f;
    float gamma = -log_approx_f(fmaxf(Bm[n] - Rm[n], 1e-6f));
    wb[n * 8 + 0] = Gm[n] - Rm[n];
    wb[n * 8 + 1] = Gm[n] - Bm[n];
    wb[n * 8 + 2] = Bm[n] - Rm[n];
    wb[n * 8 + 3] = alpha;
    wb[n * 8 + 4] = beta;
    wb[n * 8 + 5] = gamma;
  }
  flags[0] = (float)c1;
  flags[1] = (float)c3;
  flags[2] = (float)anyc;
}

__global__ __launch_bounds__(256) void wb_apply_k(
    float* __restrict__ x, const float* __restrict__ wb, const float* __restrict__ flags) {
  int idx = blockIdx.x * 256 + threadIdx.x;
  int n = idx >> 18, p = idx & (HW - 1);
  size_t bR = (size_t)(n * 3 + 0) * HW + p;
  size_t bG = bR + HW, bB = bG + HW;
  float R = x[bR], G = x[bG], B = x[bB];
  float GmR = wb[n * 8 + 0], GmB = wb[n * 8 + 1], BmR = wb[n * 8 + 2];
  float alpha = wb[n * 8 + 3], beta = wb[n * 8 + 4], gamma = wb[n * 8 + 5];
  bool c1 = flags[0] != 0.f, c3 = flags[1] != 0.f, anyc = flags[2] != 0.f;
  float B2 = c1 ? B + GmB * (1.f - B) * G * beta : B;
  float G2 = c3 ? G + BmR * (1.f - G) * B2 * gamma : G;
  float R2 = R + GmR * (1.f - R) * G2 * alpha;
  x[bR] = anyc ? satf_(R) : satf_(R2);
  x[bG] = anyc ? satf_(G) : satf_(G2);
  x[bB] = anyc ? satf_(B) : satf_(B2);
}

__global__ __launch_bounds__(256) void stretch_apply_k(
    float* __restrict__ x, const float* __restrict__ sums, const float* __restrict__ sqs) {
  int e = blockIdx.x * 256 + threadIdx.x;   // over N*3*HW
  int nc = e >> 18;
  float m = sums[nc] * (1.f / HW);
  float var = (sqs[nc] - (float)HW * m * m) * (1.f / (HW - 1));
  float s = sqrtf(fmaxf(var, 0.f));
  float lo = m - 2.f * s, hi = m + 2.f * s;
  float v = fminf(fmaxf(x[e], lo), hi);
  v = (v - lo) / (hi - lo + 1e-8f);
  x[e] = satf_(v);
}

__global__ __launch_bounds__(256) void lum_stats_k(
    const float* __restrict__ x, float* __restrict__ lsum, float* __restrict__ lsq) {
  __shared__ float ss[256], s2[256];
  int n = blockIdx.x;
  const float* R = x + (size_t)(n * 3 + 0) * HW;
  const float* G = R + HW;
  const float* B = G + HW;
  float s = 0.f, q = 0.f;
  for (int i = threadIdx.x; i < HW; i += 256) {
    float Y = 0.299f * R[i] + 0.587f * G[i] + 0.114f * B[i];
    s += Y;
    q += Y * Y;
  }
  ss[threadIdx.x] = s;
  s2[threadIdx.x] = q;
  __syncthreads();
  for (int off = 128; off > 0; off >>= 1) {
    if (threadIdx.x < off) {
      ss[threadIdx.x] += ss[threadIdx.x + off];
      s2[threadIdx.x] += s2[threadIdx.x + off];
    }
    __syncthreads();
  }
  if (threadIdx.x == 0) { lsum[n] = ss[0]; lsq[n] = s2[0]; }
}

__global__ __launch_bounds__(256) void lum_apply_k(
    float* __restrict__ x, const float* __restrict__ lsum, const float* __restrict__ lsq) {
  int idx = blockIdx.x * 256 + threadIdx.x;   // over N*HW
  int n = idx >> 18, p = idx & (HW - 1);
  float m = lsum[n] * (1.f / HW);
  float var = (lsq[n] - (float)HW * m * m) * (1.f / (HW - 1));
  float s = sqrtf(fmaxf(var, 0.f));
  float lo = m - 2.f * s, hi = m + 2.f * s;
  size_t bR = (size_t)(n * 3 + 0) * HW + p;
  size_t bG = bR + HW, bB = bG + HW;
  float R = x[bR], G = x[bG], B = x[bB];
  float Y = 0.299f * R + 0.587f * G + 0.114f * B;
  float y = fminf(fmaxf(Y, lo), hi);
  float Yo = (y - lo) / (hi - lo + 1e-8f);
  float sc = Yo / (Y + 1e-6f);
  x[bR] = satf_(R * sc);
  x[bG] = satf_(G * sc);
  x[bB] = satf_(B * sc);
}

// ---------------------------------------------------------------------------
// Launcher
// ---------------------------------------------------------------------------
extern "C" void kernel_launch(void* const* d_in, const int* in_sizes, int n_in,
                              void* d_out, int out_size, void* d_ws, size_t ws_size,
                              hipStream_t stream) {
  const float* x = (const float*)d_in[0];
  auto P = [&](int i) { return (const float*)d_in[i]; };

  // Workspace layout: three f16 8-ch activation buffers, one f32 3-ch image,
  // then the small f32 stats region.
  const size_t S8 = (size_t)NB * 8 * HW;   // elements of an 8-ch tensor
  const size_t S3 = (size_t)NB * 3 * HW;   // elements of a 3-ch tensor
  char* base = (char*)d_ws;
  _Float16* h0 = (_Float16*)base;                        // block residual
  _Float16* t1 = (_Float16*)(base + S8 * 2);             // scratch / atten out
  _Float16* t2 = (_Float16*)(base + 2 * S8 * 2);         // post-c2
  float* xbuf = (float*)(base + 3 * S8 * 2);             // 3-ch inter-stage
  float* st   = xbuf + S3;                               // stats region
  float* sums = st;                 // [64]
  float* sqs  = st + 64;            // [64]
  float* cs   = st + 128;           // [64] channel-attention scales
  float* wb   = st + 192;           // [64] wb coefficients
  float* fl   = st + 256;           // [3]  wb flags
  float* lsum = st + 272;           // [8]
  float* lsq  = st + 280;           // [8]

  const int convGrid = NB * HH * (WW / 128);   // 16384
  const int pixGrid  = NB * HW / 256;          // 8192
  const int elGrid3  = NB * 3 * HW / 256;      // 24576

  // ---- Stage 1 ----
  conv3x3_wmma_k<float, _Float16, 3, 8, 0><<<convGrid, 256, 0, stream>>>(x, P(1), P(2), nullptr, h0);
  conv3x3_wmma_k<_Float16, _Float16, 8, 8, 1><<<convGrid, 256, 0, stream>>>(h0, P(3), P(4), nullptr, t1);
  conv3x3_wmma_k<_Float16, _Float16, 8, 8, 0><<<convGrid, 256, 0, stream>>>(t1, P(5), P(6), nullptr, t2);
  reduce_stats_k<_Float16><<<NB * 8, 256, 0, stream>>>(t2, sums, sqs);
  ca_scale_k<<<NB, 32, 0, stream>>>(sums, P(7), P(8), P(9), P(10), cs);
  cp_apply_k<<<pixGrid, 256, 0, stream>>>(t2, h0, cs, P(11), P(12), P(13), P(14), t1);
  conv3x3_wmma_k<_Float16, float, 8, 3, 2><<<convGrid, 256, 0, stream>>>(t1, P(15), P(16), x, xbuf);

  // ---- Color pipeline (white balance -> channel stretch -> luminance) ----
  reduce_stats_k<float><<<NB * 3, 256, 0, stream>>>(xbuf, sums, sqs);
  wb_coeff_k<<<1, 32, 0, stream>>>(sums, wb, fl);
  wb_apply_k<<<pixGrid, 256, 0, stream>>>(xbuf, wb, fl);
  reduce_stats_k<float><<<NB * 3, 256, 0, stream>>>(xbuf, sums, sqs);
  stretch_apply_k<<<elGrid3, 256, 0, stream>>>(xbuf, sums, sqs);
  lum_stats_k<<<NB, 256, 0, stream>>>(xbuf, lsum, lsq);
  lum_apply_k<<<pixGrid, 256, 0, stream>>>(xbuf, lsum, lsq);

  // ---- Stage 2 ----
  conv3x3_wmma_k<float, _Float16, 3, 8, 0><<<convGrid, 256, 0, stream>>>(xbuf, P(17), P(18), nullptr, h0);
  conv3x3_wmma_k<_Float16, _Float16, 8, 8, 1><<<convGrid, 256, 0, stream>>>(h0, P(19), P(20), nullptr, t1);
  conv3x3_wmma_k<_Float16, _Float16, 8, 8, 0><<<convGrid, 256, 0, stream>>>(t1, P(21), P(22), nullptr, t2);
  reduce_stats_k<_Float16><<<NB * 8, 256, 0, stream>>>(t2, sums, sqs);
  ca_scale_k<<<NB, 32, 0, stream>>>(sums, P(23), P(24), P(25), P(26), cs);
  cp_apply_k<<<pixGrid, 256, 0, stream>>>(t2, h0, cs, P(27), P(28), P(29), P(30), t1);
  conv3x3_wmma_k<_Float16, float, 8, 3, 2><<<convGrid, 256, 0, stream>>>(t1, P(31), P(32), x, (float*)d_out);
}
